// CrossModalityAttention_34677565947907
// MI455X (gfx1250) — matrix operand order repaired and checked by reading, not compiled, not run on hardware
//
#include <hip/hip_runtime.h>
#include <hip/hip_bf16.h>
#include <math.h>
#include <stdint.h>

// ---------------- types ----------------
typedef __bf16 bf16_t;
typedef __attribute__((ext_vector_type(16))) __bf16 v16bf;
typedef __attribute__((ext_vector_type(8)))  __bf16 v8bf;
typedef __attribute__((ext_vector_type(8)))  float  v8f;

// ---------------- problem constants ----------------
#define NB     8
#define LQ     2048
#define LK     2048
#define HID    1024
#define IMGD   1024
#define TXTD   768

// ---------------- CDNA5 async global->LDS copy (ASYNCcnt path) ----------------
// Moves 16 bytes per enabled lane directly into LDS without a VGPR round trip.
static __device__ __forceinline__ void async_cp16(void* ldsDst, const void* gSrc) {
  asm volatile("global_load_async_to_lds_b128 %0, %1, off"
               :: "v"((uint32_t)(uintptr_t)ldsDst), "v"(gSrc)
               : "memory");
}
static __device__ __forceinline__ void wait_async0() {
  asm volatile("s_wait_asynccnt 0x0" ::: "memory");
}

// ---------------- fragment loaders (CDNA5 16-bit WMMA layouts) ----------------
// A 16x32 (bf16): lane = row (lane&15); elem e<8 -> k = e + 8*hi ; e>=8 -> k = (e-8)+16+8*hi
static __device__ __forceinline__ v16bf load_a_frag(const bf16_t* rowPtr, int hiH) {
  v8bf lo = *(const v8bf*)(rowPtr + 8 * hiH);
  v8bf hi = *(const v8bf*)(rowPtr + 16 + 8 * hiH);
  return __builtin_shufflevector(lo, hi, 0,1,2,3,4,5,6,7,8,9,10,11,12,13,14,15);
}
// B 32x16 (bf16): lane = col (lane&15); elem e<8 -> k = e + 8*hi ; e>=8 -> k = (e-8)+16+8*hi
static __device__ __forceinline__ v16bf load_b_frag(const bf16_t* colPtr, int hiH) {
  const bf16_t* p = colPtr + 8 * hiH;
  v8bf lo = *(const v8bf*)p;
  v8bf hi = *(const v8bf*)(p + 16);
  return __builtin_shufflevector(lo, hi, 0,1,2,3,4,5,6,7,8,9,10,11,12,13,14,15);
}

// ---------------- f32 -> bf16 convert (vectorized) ----------------
__global__ void cma_cvt_bf16(const float* __restrict__ in, bf16_t* __restrict__ out, int n4) {
  int i = blockIdx.x * blockDim.x + threadIdx.x;
  int stride = gridDim.x * blockDim.x;
  for (; i < n4; i += stride) {
    float4 v = ((const float4*)in)[i];
    union { uint64_t u; bf16_t h[4]; } r;
    r.h[0] = (bf16_t)v.x; r.h[1] = (bf16_t)v.y; r.h[2] = (bf16_t)v.z; r.h[3] = (bf16_t)v.w;
    ((uint64_t*)out)[i] = r.u;
  }
}

// ---------------- W[K][1024] f32 -> Wt[1024][K] bf16 ----------------
__global__ void cma_wtr(const float* __restrict__ W, bf16_t* __restrict__ Wt, int Kd) {
  int total = Kd * 1024;
  for (int i = blockIdx.x * blockDim.x + threadIdx.x; i < total; i += gridDim.x * blockDim.x) {
    int k = i >> 10, n = i & 1023;
    Wt[(size_t)n * Kd + k] = (bf16_t)W[i];
  }
}

// ---------------- async tile staging for the projection GEMM ----------------
static __device__ __forceinline__ void stage_tile(const bf16_t* __restrict__ X,
                                                  const bf16_t* __restrict__ Wt,
                                                  bf16_t* xbuf, bf16_t* wbuf,
                                                  int Kd, int m0, int n0, int kt,
                                                  int row_s, int seg) {
  const bf16_t* xs = X + (size_t)(m0 + row_s) * Kd + kt + seg * 16;
  bf16_t* xd = xbuf + row_s * 40 + seg * 16;
  async_cp16(xd, xs);
  async_cp16(xd + 8, xs + 8);
  const bf16_t* wsv = Wt + (size_t)(n0 + row_s) * Kd + kt + seg * 16;
  bf16_t* wd = wbuf + row_s * 40 + seg * 16;
  async_cp16(wd, wsv);
  async_cp16(wd + 8, wsv + 8);
}

// ---------------- projection GEMM: out[m][n] = X[m][:]·Wt[n][:] + bias[n] ----------------
// M = 16384, N = 1024. vmode==1 writes transposed per-batch: Vt[b][n][key]
__global__ __launch_bounds__(256) void cma_proj(const bf16_t* __restrict__ X,
                                                const bf16_t* __restrict__ Wt,
                                                const float*  __restrict__ bias,
                                                bf16_t* __restrict__ out,
                                                int Kd, int vmode) {
  __shared__ bf16_t Xs[2][128 * 40];  // double-buffered, stride 40 bf16 (80B: aligned, bank-clean)
  __shared__ bf16_t Ws[2][128 * 40];
  const int tid  = threadIdx.x;
  const int lane = tid & 31, w = tid >> 5;
  const int ln = lane & 15, hiH = lane >> 4;
  const int wm = w >> 1, wn = w & 1;           // 4x2 wave grid, each wave 32m x 64n
  const int m0 = blockIdx.x * 128, n0 = blockIdx.y * 128;

  v8f acc[2][4];
  #pragma unroll
  for (int nt = 0; nt < 4; ++nt) {
    float bv = bias[n0 + 64 * wn + 16 * nt + ln];
    v8f c = {bv, bv, bv, bv, bv, bv, bv, bv};
    acc[0][nt] = c; acc[1][nt] = c;
  }

  const int row_s = tid >> 1, seg = tid & 1;   // 2 threads per row, 32B each
  const int nsteps = Kd / 32;

  stage_tile(X, Wt, Xs[0], Ws[0], Kd, m0, n0, 0, row_s, seg);

  for (int s = 0; s < nsteps; ++s) {
    wait_async0();
    __syncthreads();
    if (s + 1 < nsteps)   // prefetch next tile into the other buffer (overlaps WMMA)
      stage_tile(X, Wt, Xs[(s + 1) & 1], Ws[(s + 1) & 1], Kd, m0, n0, (s + 1) * 32, row_s, seg);

    const int cur = s & 1;
    v16bf a[2];
    #pragma unroll
    for (int mt = 0; mt < 2; ++mt)
      a[mt] = load_a_frag(&Xs[cur][(ln + 16 * mt + 32 * wm) * 40], hiH);
    #pragma unroll
    for (int nt = 0; nt < 4; ++nt) {
      v16bf bfr = load_b_frag(&Ws[cur][(ln + 16 * nt + 64 * wn) * 40], hiH);
      #pragma unroll
      for (int mt = 0; mt < 2; ++mt)
        acc[mt][nt] = __builtin_amdgcn_wmma_f32_16x16x32_bf16(
            false, a[mt], false, bfr, (short)0, acc[mt][nt], false, false);
    }
  }

  if (vmode == 0) {
    #pragma unroll
    for (int mt = 0; mt < 2; ++mt)
      #pragma unroll
      for (int nt = 0; nt < 4; ++nt) {
        int cg = n0 + 64 * wn + 16 * nt + ln;
        #pragma unroll
        for (int r = 0; r < 8; ++r) {
          int rg = m0 + 32 * wm + 16 * mt + 8 * hiH + r;
          out[(size_t)rg * HID + cg] = (bf16_t)acc[mt][nt][r];
        }
      }
  } else {
    // transposed store: Vt[b][dim][key]; 8 consecutive keys per lane -> b128 store
    int bidx = m0 >> 11;
    #pragma unroll
    for (int mt = 0; mt < 2; ++mt)
      #pragma unroll
      for (int nt = 0; nt < 4; ++nt) {
        union { uint4 u; bf16_t h[8]; } pk;
        #pragma unroll
        for (int r = 0; r < 8; ++r) pk.h[r] = (bf16_t)acc[mt][nt][r];
        int dim = n0 + 64 * wn + 16 * nt + ln;
        int key = (m0 & 2047) + 32 * wm + 16 * mt + 8 * hiH;
        *(uint4*)(out + ((size_t)bidx * HID + dim) * LK + key) = pk.u;
      }
  }
}

// ---------------- flash attention: 1 WG = 32 query rows, 8 waves split HID ----------------
__global__ __launch_bounds__(256) void cma_attn(const bf16_t* __restrict__ Qg,
                                                const bf16_t* __restrict__ Kg,
                                                const bf16_t* __restrict__ Vtg,
                                                float* __restrict__ outp) {
  __shared__ float  Sv[32 * 33];   // score tile, stride 33 f32
  __shared__ bf16_t Pt[32 * 40];   // exp(S-m) tile, stride 40 bf16
  __shared__ float  mrow[32], lrow[32], arow[32];

  const int tid  = threadIdx.x;
  const int lane = tid & 31, w = tid >> 5;     // wave w owns dims [128w, 128w+128)
  const int ln = lane & 15, hiH = lane >> 4;
  const int b  = blockIdx.x >> 6;
  const int q0 = (blockIdx.x & 63) * 32;
  const float scale = 0.03125f;                // 1/sqrt(1024)

  if (tid < 32) { mrow[tid] = -3.0e38f; lrow[tid] = 0.0f; }

  // Q fragments register-resident: rows ln+16mt, dims 128w + 32kk
  v16bf qf[2][4];
  #pragma unroll
  for (int mt = 0; mt < 2; ++mt)
    #pragma unroll
    for (int kk = 0; kk < 4; ++kk) {
      const bf16_t* p = Qg + ((size_t)(b * LQ + q0 + ln + 16 * mt)) * HID + w * 128 + kk * 32;
      qf[mt][kk] = load_a_frag(p, hiH);
    }

  v8f o[2][8];
  v8f vz = {0.f, 0.f, 0.f, 0.f, 0.f, 0.f, 0.f, 0.f};
  #pragma unroll
  for (int mt = 0; mt < 2; ++mt)
    #pragma unroll
    for (int nt = 0; nt < 8; ++nt) o[mt][nt] = vz;

  // softmax lane mapping: 8 threads per query row, 4 columns each
  const int sr = tid >> 3;          // row 0..31
  const int sc = (tid & 7) * 4;     // column base

  for (int kb = 0; kb < LK / 32; ++kb) {
    const int k0 = kb * 32;
    for (int i = tid; i < 32 * 33; i += 256) Sv[i] = 0.0f;
    __syncthreads();

    // warm L2 for the next key block (global_prefetch_b8)
    if (kb + 1 < LK / 32) {
      const bf16_t* kpn = Kg + ((size_t)(b * LK + k0 + 32 + ln)) * HID + w * 128;
      __builtin_prefetch(kpn, 0, 3);
      __builtin_prefetch(kpn + (size_t)16 * HID, 0, 3);
      const bf16_t* vpn = Vtg + ((size_t)(b * HID + w * 128 + ln)) * LK + k0 + 32;
      __builtin_prefetch(vpn, 0, 3);
    }

    // partial S = Q·K^T over this wave's 128-dim slice
    v8f sacc[2][2];
    sacc[0][0] = vz; sacc[0][1] = vz; sacc[1][0] = vz; sacc[1][1] = vz;
    #pragma unroll
    for (int kk = 0; kk < 4; ++kk)
      #pragma unroll
      for (int nt = 0; nt < 2; ++nt) {
        const bf16_t* kp = Kg + ((size_t)(b * LK + k0 + 16 * nt + ln)) * HID + w * 128 + kk * 32;
        v16bf kf = load_b_frag(kp, hiH);
        #pragma unroll
        for (int mt = 0; mt < 2; ++mt)
          sacc[mt][nt] = __builtin_amdgcn_wmma_f32_16x16x32_bf16(
              false, qf[mt][kk], false, kf, (short)0, sacc[mt][nt], false, false);
      }
    // cross-wave reduction via LDS float atomics
    #pragma unroll
    for (int mt = 0; mt < 2; ++mt)
      #pragma unroll
      for (int nt = 0; nt < 2; ++nt)
        #pragma unroll
        for (int r = 0; r < 8; ++r)
          atomicAdd(&Sv[(16 * mt + 8 * hiH + r) * 33 + 16 * nt + ln], sacc[mt][nt][r]);
    __syncthreads();

    // online softmax update: all 256 threads, butterfly reductions over 8 lanes/row
    {
      float mold = mrow[sr];
      float s0 = Sv[sr * 33 + sc + 0] * scale;
      float s1 = Sv[sr * 33 + sc + 1] * scale;
      float s2 = Sv[sr * 33 + sc + 2] * scale;
      float s3 = Sv[sr * 33 + sc + 3] * scale;
      float mx = fmaxf(fmaxf(fmaxf(s0, s1), fmaxf(s2, s3)), mold);
      #pragma unroll
      for (int off = 1; off < 8; off <<= 1)
        mx = fmaxf(mx, __shfl_xor(mx, off, 8));
      float p0 = __expf(s0 - mx), p1 = __expf(s1 - mx);
      float p2 = __expf(s2 - mx), p3 = __expf(s3 - mx);
      Pt[sr * 40 + sc + 0] = (bf16_t)p0;
      Pt[sr * 40 + sc + 1] = (bf16_t)p1;
      Pt[sr * 40 + sc + 2] = (bf16_t)p2;
      Pt[sr * 40 + sc + 3] = (bf16_t)p3;
      float sum = (p0 + p1) + (p2 + p3);
      #pragma unroll
      for (int off = 1; off < 8; off <<= 1)
        sum += __shfl_xor(sum, off, 8);
      if ((tid & 7) == 0) {
        float alpha = __expf(mold - mx);
        lrow[sr] = lrow[sr] * alpha + sum;
        arow[sr] = alpha;
        mrow[sr] = mx;
      }
    }
    __syncthreads();

    // rescale running output, then O += P·V (V^T read directly from L2)
    float al[2][8];
    #pragma unroll
    for (int mt = 0; mt < 2; ++mt)
      #pragma unroll
      for (int r = 0; r < 8; ++r) al[mt][r] = arow[16 * mt + 8 * hiH + r];
    #pragma unroll
    for (int mt = 0; mt < 2; ++mt)
      #pragma unroll
      for (int nt = 0; nt < 8; ++nt)
        #pragma unroll
        for (int r = 0; r < 8; ++r) o[mt][nt][r] *= al[mt][r];

    v16bf pf[2];
    #pragma unroll
    for (int mt = 0; mt < 2; ++mt)
      pf[mt] = load_a_frag(Pt + (ln + 16 * mt) * 40, hiH);
    #pragma unroll
    for (int nt = 0; nt < 8; ++nt) {
      const bf16_t* vp = Vtg + ((size_t)(b * HID + w * 128 + 16 * nt + ln)) * LK + k0;
      v16bf vf = load_b_frag(vp, hiH);
      #pragma unroll
      for (int mt = 0; mt < 2; ++mt)
        o[mt][nt] = __builtin_amdgcn_wmma_f32_16x16x32_bf16(
            false, pf[mt], false, vf, (short)0, o[mt][nt], false, false);
    }
  }
  __syncthreads();

  float li[2][8];
  #pragma unroll
  for (int mt = 0; mt < 2; ++mt)
    #pragma unroll
    for (int r = 0; r < 8; ++r) li[mt][r] = 1.0f / lrow[16 * mt + 8 * hiH + r];
  #pragma unroll
  for (int mt = 0; mt < 2; ++mt)
    #pragma unroll
    for (int nt = 0; nt < 8; ++nt)
      #pragma unroll
      for (int r = 0; r < 8; ++r)
        outp[((size_t)(b * LQ + q0 + 16 * mt + 8 * hiH + r)) * HID + w * 128 + 16 * nt + ln] =
            o[mt][nt][r] * li[mt][r];
}

// ---------------- host launcher ----------------
extern "C" void kernel_launch(void* const* d_in, const int* in_sizes, int n_in,
                              void* d_out, int out_size, void* d_ws, size_t ws_size,
                              hipStream_t stream) {
  const float* img = (const float*)d_in[0];
  const float* txt = (const float*)d_in[1];
  const float* Wq  = (const float*)d_in[2];
  const float* bq  = (const float*)d_in[3];
  const float* Wk  = (const float*)d_in[4];
  const float* bk  = (const float*)d_in[5];
  const float* Wv  = (const float*)d_in[6];
  const float* bv  = (const float*)d_in[7];
  float* out = (float*)d_out;

  char* ws = (char*)d_ws;
  const size_t SZ_XIMG = (size_t)NB * LQ * IMGD * 2;
  const size_t SZ_XTXT = (size_t)NB * LK * TXTD * 2;
  const size_t SZ_WTQ  = (size_t)IMGD * HID * 2;
  const size_t SZ_WTK  = (size_t)TXTD * HID * 2;
  const size_t SZ_QKV  = (size_t)NB * LQ * HID * 2;
  size_t off = 0;
  bf16_t* Ximg = (bf16_t*)(ws + off); off += SZ_XIMG;
  bf16_t* Xtxt = (bf16_t*)(ws + off); off += SZ_XTXT;
  bf16_t* WtQ  = (bf16_t*)(ws + off); off += SZ_WTQ;
  bf16_t* WtK  = (bf16_t*)(ws + off); off += SZ_WTK;
  bf16_t* WtV  = (bf16_t*)(ws + off); off += SZ_WTK;
  bf16_t* Qb   = (bf16_t*)(ws + off); off += SZ_QKV;
  bf16_t* Kb   = (bf16_t*)(ws + off); off += SZ_QKV;
  bf16_t* Vt   = (bf16_t*)(ws + off); off += SZ_QKV;

  // 1) convert activations to bf16
  {
    int n4 = NB * LQ * IMGD / 4;
    cma_cvt_bf16<<<(n4 + 255) / 256, 256, 0, stream>>>(img, Ximg, n4);
    n4 = NB * LK * TXTD / 4;
    cma_cvt_bf16<<<(n4 + 255) / 256, 256, 0, stream>>>(txt, Xtxt, n4);
  }
  // 2) transpose+convert weights
  cma_wtr<<<(IMGD * HID + 255) / 256, 256, 0, stream>>>(Wq, WtQ, IMGD);
  cma_wtr<<<(TXTD * HID + 255) / 256, 256, 0, stream>>>(Wk, WtK, TXTD);
  cma_wtr<<<(TXTD * HID + 255) / 256, 256, 0, stream>>>(Wv, WtV, TXTD);

  // 3) projections (WMMA GEMMs, async double-buffered staging)
  dim3 pg(NB * LQ / 128, HID / 128);
  cma_proj<<<pg, 256, 0, stream>>>(Ximg, WtQ, bq, Qb, IMGD, 0);
  cma_proj<<<pg, 256, 0, stream>>>(Xtxt, WtK, bk, Kb, TXTD, 0);
  cma_proj<<<pg, 256, 0, stream>>>(Xtxt, WtV, bv, Vt, TXTD, 1);  // transposed V

  // 4) flash attention
  cma_attn<<<NB * (LQ / 32), 256, 0, stream>>>(Qb, Kb, Vt, out);
}
